// ChronoSynth_43370579755409
// MI455X (gfx1250) — matrix operand order repaired and checked by reading, not compiled, not run on hardware
//
#include <hip/hip_runtime.h>

typedef __attribute__((ext_vector_type(16))) _Float16 v16h;
typedef __attribute__((ext_vector_type(8)))  _Float16 v8h;
typedef __attribute__((ext_vector_type(8)))  float    v8f;

// ---------------------------------------------------------------------------
// Weight packing: fp32 [Cout][K] -> f16 [Cout][Kp], zero padded K..Kp
// ---------------------------------------------------------------------------
__global__ void pack_w_k(const float* __restrict__ src, _Float16* __restrict__ dst,
                         int Cout, int K, int Kp) {
  int i = blockIdx.x * blockDim.x + threadIdx.x;
  if (i >= Cout * Kp) return;
  int m = i / Kp, k = i - m * Kp;
  dst[i] = (k < K) ? (_Float16)src[m * K + k] : (_Float16)0.0f;
}

// fuse 1x1: src [65][260] with K-order ch*4+t  ->  dst [65][288] with K-order t*65+ch
__global__ void pack_fuse_k(const float* __restrict__ src, _Float16* __restrict__ dst) {
  int i = blockIdx.x * blockDim.x + threadIdx.x;
  if (i >= 65 * 288) return;
  int m = i / 288, k = i - m * 288;
  _Float16 v = (_Float16)0.0f;
  if (k < 260) { int t = k / 65, ch = k - t * 65; v = (_Float16)src[m * 260 + ch * 4 + t]; }
  dst[i] = v;
}

// ---------------------------------------------------------------------------
// Build fbt f16 [B=2][T=4][65][128][128]: features + |ctx[t]-tf| time channel
// ---------------------------------------------------------------------------
__global__ void build_fbt_k(const float* __restrict__ feat, const float* __restrict__ tptr,
                            _Float16* __restrict__ fbt) {
  const int HW = 128 * 128;
  int i = blockIdx.x * blockDim.x + threadIdx.x;
  if (i >= 8 * 65 * HW) return;
  int pix = i % HW;
  int c = (i / HW) % 65;
  int t = (i / (HW * 65)) % 4;
  int b = i / (HW * 65 * 4);
  float v;
  if (c < 64) v = feat[(((size_t)(b * 64 + c) * 4 + t) * HW) + pix];
  else        v = fabsf((float)(t - 1) - tptr[0]);   // ctx = [-1,0,1,2]
  fbt[i] = (_Float16)v;
}

// ---------------------------------------------------------------------------
// LDS-staged implicit-GEMM conv via WMMA (f32 += f16 x f16).
// MODE 0: 3x3 stride1 pad1.  MODE 1: 1x1 (async global->LDS staging).
// Block = 256 threads = 8 waves = 128 pixels of one output row.
// Each wave: 16-pixel N tile, loops all Cout 16-tiles (<=5) sharing one B build.
// All guarded accesses are branchless: clamped address + v_cndmask select.
// ---------------------------------------------------------------------------
template <int MODE>
__global__ __launch_bounds__(256)
void conv_lds_k(const _Float16* __restrict__ act, const _Float16* __restrict__ wgt,
                const float* __restrict__ bias, _Float16* __restrict__ out,
                int Nimg, int Cin, int Hin, int Win, int Cout, int Hout, int Wout,
                int Kp, int Ktot, int lrelu) {
  extern __shared__ _Float16 patch[];
  const int tid    = threadIdx.x;
  const int Wtiles = Wout >> 7;
  int bid = blockIdx.x;
  int wt  = bid % Wtiles;
  int y   = (bid / Wtiles) % Hout;
  int img = bid / (Wtiles * Hout);
  int x0  = wt << 7;

  const _Float16* ibase = act + (size_t)img * Cin * Hin * Win;

  if (MODE == 0) {
    // ---- interior: coalesced b32 global loads -> ds_store_b32 (branchless) ----
    int total = Cin * 3 * 64;                       // u32 pairs per (cin,row)
    for (int t = tid; t < total; t += 256) {
      int cin = t / 192;
      int rr  = t - cin * 192;
      int r   = rr >> 6;
      int q   = rr & 63;
      int gy  = y + r - 1;
      bool ok = (unsigned)gy < (unsigned)Hin;
      int gyc = min(max(gy, 0), Hin - 1);
      unsigned v = *(const unsigned*)(ibase + ((size_t)cin * Hin + gyc) * Win + x0 + 2 * q);
      v = ok ? v : 0u;
      *(unsigned*)(patch + cin * 396 + r * 132 + 2 + 2 * q) = v;
    }
    // ---- halo columns (logical -1 and 128 -> phys 1 and 130), branchless ----
    int totalh = Cin * 3 * 2;
    for (int t = tid; t < totalh; t += 256) {
      int cin  = t / 6;
      int rr   = t - cin * 6;
      int r    = rr >> 1;
      int side = rr & 1;
      int gy   = y + r - 1;
      int gx   = side ? (x0 + 128) : (x0 - 1);
      bool ok  = ((unsigned)gy < (unsigned)Hin) && ((unsigned)gx < (unsigned)Win);
      int gyc  = min(max(gy, 0), Hin - 1);
      int gxc  = min(max(gx, 0), Win - 1);
      _Float16 v = ibase[((size_t)cin * Hin + gyc) * Win + gxc];
      patch[cin * 396 + r * 132 + (side ? 130 : 1)] = ok ? v : (_Float16)0.0f;
    }
  } else {
    // ---- 1x1: async DMA rows [Cin][128] straight into LDS ----
    unsigned pbase = (unsigned)(unsigned long long)(size_t)patch;
    unsigned long long gb = (unsigned long long)(size_t)ibase;
    int total = Cin * 64;                           // u32 chunks
    for (int t = tid; t < total; t += 256) {
      int cin = t >> 6;
      int q   = t & 63;
      unsigned goff = (unsigned)(((((unsigned)cin * Hin + y) * Win) + 2 * q) * 2);
      unsigned loff = pbase + (unsigned)((cin * 128 + 2 * q) * 2);
      asm volatile("global_load_async_to_lds_b32 %0, %1, %2"
                   :: "v"(loff), "v"(goff), "s"(gb) : "memory");
    }
    asm volatile("s_wait_asynccnt 0x0" ::: "memory");
  }
  __syncthreads();

  // ---------------- compute ----------------
  int wslot = tid >> 5;
  int lane  = tid & 31;
  int hi    = lane >> 4;
  int ln    = lane & 15;
  int p     = (wslot << 4) + ln;                    // pixel column within 128-tile

  int Mtiles = (Cout + 15) >> 4;                    // <= 5
  v8f acc[5];
#pragma unroll
  for (int mt = 0; mt < 5; ++mt) { v8f z = {}; acc[mt] = z; }

  for (int kb = 0; kb < Ktot; kb += 32) {
    // ---- B from LDS: element j holds K = kb + j + hi*16, column = p ----
    v16h b;
#pragma unroll
    for (int j = 0; j < 16; ++j) {
      int k  = kb + j + (hi << 4);
      int kc = min(k, Ktot - 1);                    // clamped, always in-patch
      _Float16 v;
      if (MODE == 0) {
        int cin = kc / 9, tap = kc - cin * 9;
        v = patch[cin * 396 + (tap / 3) * 132 + p + (tap % 3) + 1];
      } else {
        v = patch[kc * 128 + p];
      }
      b[j] = (k < Ktot) ? v : (_Float16)0.0f;
    }
    // ---- A per Cout tile: two 16B vector loads, then WMMA ----
#pragma unroll
    for (int mt = 0; mt < 5; ++mt) {
      if (mt < Mtiles) {
        int arow = mt * 16 + ln; if (arow > Cout - 1) arow = Cout - 1;
        const _Float16* ab = wgt + (size_t)arow * Kp + kb + hi * 16;
        const v8h w0 = *(const v8h*)ab;
        const v8h w1 = *(const v8h*)(ab + 8);
        v16h a;
#pragma unroll
        for (int j = 0; j < 8; ++j) { a[j] = w0[j]; a[j + 8] = w1[j]; }
        acc[mt] = __builtin_amdgcn_wmma_f32_16x16x32_f16(false, a, false, b,
                                                         (short)0, acc[mt],
                                                         false, false);
      }
    }
  }

  // ---- bias + lrelu + store ----
  int HWo  = Hout * Wout;
  int pixg = y * Wout + x0 + p;
  _Float16* obase = out + (size_t)img * Cout * HWo + pixg;
#pragma unroll
  for (int mt = 0; mt < 5; ++mt) {
    if (mt < Mtiles) {
#pragma unroll
      for (int r = 0; r < 8; ++r) {
        int m = mt * 16 + r + hi * 8;
        if (m < Cout) {
          float v = acc[mt][r] + bias[m];
          if (lrelu) v = (v >= 0.0f) ? v : 0.2f * v;
          obase[(size_t)m * HWo] = (_Float16)v;
        }
      }
    }
  }
}

// ---------------------------------------------------------------------------
// convT k3 s2 p1 op1 (lhs-dilated conv, pad (1,2)) — direct-global WMMA.
// B gather is branchless: clamped address + single-predicate select.
// ---------------------------------------------------------------------------
__global__ __launch_bounds__(256)
void convt_wmma_k(const _Float16* __restrict__ act, const _Float16* __restrict__ wgt,
                  const float* __restrict__ bias, _Float16* __restrict__ out,
                  int Nimg, int Cin, int Hin, int Win, int Cout, int Hout, int Wout,
                  int Kp, int Ktot, int lrelu) {
  int gwave = (blockIdx.x * blockDim.x + threadIdx.x) >> 5;
  int lane  = threadIdx.x & 31;
  int hi    = lane >> 4;
  int ln    = lane & 15;

  int HWo       = Hout * Wout;
  int PtilesImg = HWo >> 4;
  int Ptiles    = Nimg * PtilesImg;
  int Mtiles    = (Cout + 15) >> 4;
  if (gwave >= Mtiles * Ptiles) return;

  int mt  = gwave / Ptiles;
  int pt  = gwave - mt * Ptiles;
  int img = pt / PtilesImg;
  int pin = (pt - img * PtilesImg) << 4;
  int py  = pin / Wout;
  int px0 = pin - py * Wout;
  int x   = px0 + ln;

  int arow = mt * 16 + ln; if (arow > Cout - 1) arow = Cout - 1;
  const _Float16* abase = wgt + (size_t)arow * Kp;
  const _Float16* ibase = act + (size_t)img * Cin * Hin * Win;

  v8f c = {};
  for (int kb = 0; kb < Ktot; kb += 32) {
    v16h a;
    {
      const v8h w0 = *(const v8h*)(abase + kb + hi * 16);
      const v8h w1 = *(const v8h*)(abase + kb + hi * 16 + 8);
#pragma unroll
      for (int j = 0; j < 8; ++j) { a[j] = w0[j]; a[j + 8] = w1[j]; }
    }
    v16h b;
#pragma unroll
    for (int j = 0; j < 16; ++j) {
      int k  = kb + j + (hi << 4);
      int kc = min(k, Ktot - 1);
      int cin = kc / 9, tap = kc - cin * 9;
      int dy = py + tap / 3 - 1;
      int dx = x + tap % 3 - 1;
      int iy = dy >> 1, ix = dx >> 1;
      bool ok = (k < Ktot) && (dy >= 0) && (dx >= 0) && !(dy & 1) && !(dx & 1) &&
                (iy < Hin) && (ix < Win);
      int iyc = min(max(iy, 0), Hin - 1);
      int ixc = min(max(ix, 0), Win - 1);
      _Float16 v = ibase[((size_t)cin * Hin + iyc) * Win + ixc];
      b[j] = ok ? v : (_Float16)0.0f;
    }
    c = __builtin_amdgcn_wmma_f32_16x16x32_f16(false, a, false, b, (short)0, c,
                                               false, false);
  }

  int pix = pin + ln;
  _Float16* obase = out + (size_t)img * Cout * HWo + pix;
#pragma unroll
  for (int r = 0; r < 8; ++r) {
    int m = mt * 16 + r + hi * 8;
    if (m < Cout) {
      float v = c[r] + bias[m];
      if (lrelu) v = (v >= 0.0f) ? v : 0.2f * v;
      obase[(size_t)m * HWo] = (_Float16)v;
    }
  }
}

// ---------------------------------------------------------------------------
// Channel softmax, layout [NI][C][HW], in place
// ---------------------------------------------------------------------------
template <int C>
__global__ void softmax_ch_k(_Float16* __restrict__ data, int NI, int HW) {
  int i = blockIdx.x * blockDim.x + threadIdx.x;
  if (i >= NI * HW) return;
  int img = i / HW, pix = i - img * HW;
  _Float16* p = data + (size_t)img * C * HW + pix;
  float v[C];
  float mx = -1e30f;
#pragma unroll
  for (int c = 0; c < C; ++c) { v[c] = (float)p[(size_t)c * HW]; mx = fmaxf(mx, v[c]); }
  float s = 0.0f;
#pragma unroll
  for (int c = 0; c < C; ++c) { v[c] = __expf(v[c] - mx); s += v[c]; }
  float inv = 1.0f / s;
#pragma unroll
  for (int c = 0; c < C; ++c) p[(size_t)c * HW] = (_Float16)(v[c] * inv);
}

// ---------------------------------------------------------------------------
// AdaCoF synthesis + occlusion blend.
// ---------------------------------------------------------------------------
__global__ __launch_bounds__(256)
void synth_k(const _Float16* __restrict__ Wg, const _Float16* __restrict__ Al,
             const _Float16* __restrict__ Be, const _Float16* __restrict__ Oc,
             const float* __restrict__ frames, float* __restrict__ out) {
  const int W = 256, HW = 65536;
  int i = blockIdx.x * blockDim.x + threadIdx.x;
  if (i >= 2 * HW) return;
  int b = i / HW, pix = i - b * HW;
  int y = pix >> 8, x = pix & 255;
  float acc0 = 0.f, acc1 = 0.f, acc2 = 0.f;
  for (int t = 0; t < 4; ++t) {
    float occ = (float)Oc[(size_t)(b * 4 + t) * HW + pix];
    const _Float16* wp = Wg + (size_t)(b * 4 + t) * 25 * HW + pix;
    const _Float16* ap = Al + (size_t)(b * 4 + t) * 25 * HW + pix;
    const _Float16* bp = Be + (size_t)(b * 4 + t) * 25 * HW + pix;
    const float* fr = frames + (size_t)(t * 2 + b) * 3 * HW;
#pragma unroll
    for (int k = 0; k < 25; ++k) {
      float w  = (float)wp[(size_t)k * HW] * occ;
      float al = (float)ap[(size_t)k * HW];
      float be = (float)bp[(size_t)k * HW];
      float pyf = (float)(y + k / 5) + al;
      float pxf = (float)(x + k % 5) + be;
      float fy = floorf(pyf), fx = floorf(pxf);
      float ry = pyf - fy,    rx = pxf - fx;
      int iy0 = (int)fy, ix0 = (int)fx;
      int y0 = min(max(iy0, 0), 259), y1 = min(max(iy0 + 1, 0), 259);
      int x0 = min(max(ix0, 0), 259), x1 = min(max(ix0 + 1, 0), 259);
      y0 = min(max(y0 - 2, 0), 255); y1 = min(max(y1 - 2, 0), 255);
      x0 = min(max(x0 - 2, 0), 255); x1 = min(max(x1 - 2, 0), 255);
      float w00 = w * (1.f - ry) * (1.f - rx);
      float w01 = w * (1.f - ry) * rx;
      float w10 = w * ry * (1.f - rx);
      float w11 = w * ry * rx;
      int o00 = y0 * W + x0, o01 = y0 * W + x1, o10 = y1 * W + x0, o11 = y1 * W + x1;
      acc0 += w00 * fr[o00]          + w01 * fr[o01]          + w10 * fr[o10]          + w11 * fr[o11];
      acc1 += w00 * fr[HW + o00]     + w01 * fr[HW + o01]     + w10 * fr[HW + o10]     + w11 * fr[HW + o11];
      acc2 += w00 * fr[2 * HW + o00] + w01 * fr[2 * HW + o01] + w10 * fr[2 * HW + o10] + w11 * fr[2 * HW + o11];
    }
  }
  out[(size_t)(b * 3 + 0) * HW + pix] = acc0;
  out[(size_t)(b * 3 + 1) * HW + pix] = acc1;
  out[(size_t)(b * 3 + 2) * HW + pix] = acc2;
}

// ---------------------------------------------------------------------------
extern "C" void kernel_launch(void* const* d_in, const int* in_sizes, int n_in,
                              void* d_out, int out_size, void* d_ws, size_t ws_size,
                              hipStream_t stream) {
  (void)in_sizes; (void)n_in; (void)out_size; (void)ws_size;
  const float* features = (const float*)d_in[0];
  const float* frames   = (const float*)d_in[1];
  const float* tptr     = (const float*)d_in[4];

  _Float16* ws = (_Float16*)d_ws;
  size_t off = 0;
  auto alloc = [&](size_t n) { size_t o = off; off += (n + 127) & ~(size_t)127; return o; };

  // ---- weight slots: 3 subnets x {w1,w2,wt,w3}, occ x4, fuse ----
  int widx[16], bidx[16], wCout[16], wK[16], wKp[16];
  size_t woff[16];
  int s = 0;
  for (int p = 0; p < 3; ++p) {
    int base = 5 + p * 8;
    widx[s]=base+0; bidx[s]=base+1; wCout[s]=64; wK[s]=585; wKp[s]=608; ++s; // w1 65->64
    widx[s]=base+2; bidx[s]=base+3; wCout[s]=25; wK[s]=576; wKp[s]=576; ++s; // w2 64->25
    widx[s]=base+4; bidx[s]=base+5; wCout[s]=25; wK[s]=225; wKp[s]=256; ++s; // wt 25->25
    widx[s]=base+6; bidx[s]=base+7; wCout[s]=25; wK[s]=225; wKp[s]=256; ++s; // w3 25->25
  }
  widx[s]=29; bidx[s]=30; wCout[s]=64; wK[s]=585; wKp[s]=608; ++s;  // occ w1
  widx[s]=31; bidx[s]=32; wCout[s]=64; wK[s]=576; wKp[s]=576; ++s;  // occ w2
  widx[s]=33; bidx[s]=34; wCout[s]=64; wK[s]=576; wKp[s]=576; ++s;  // occ wt
  widx[s]=35; bidx[s]=36; wCout[s]=4;  wK[s]=576; wKp[s]=576; ++s;  // occ w3
  for (int j = 0; j < 16; ++j) woff[j] = alloc((size_t)wCout[j] * wKp[j]);
  size_t oFUSE = alloc(65 * 288);

  // ---- activation arena (f16 elements) ----
  const size_t HW1 = 128 * 128, HW2 = 256 * 256;
  size_t oFBT = alloc(8ull * 65 * HW1);
  size_t oOCF = alloc(2ull * 65 * HW1);
  size_t oT1  = alloc(8ull * 64 * HW1);
  size_t oT2  = alloc(8ull * 25 * HW1);
  size_t oT3  = alloc(8ull * 25 * HW2);
  size_t oBIG[3] = { alloc(2ull * 4 * 25 * HW2), alloc(2ull * 4 * 25 * HW2),
                     alloc(2ull * 4 * 25 * HW2) };
  size_t oOCL = alloc(2ull * 4 * HW2);

  // ---- pack weights ----
  for (int j = 0; j < 16; ++j) {
    int n = wCout[j] * wKp[j];
    pack_w_k<<<(n + 255) / 256, 256, 0, stream>>>((const float*)d_in[widx[j]],
                                                  ws + woff[j], wCout[j], wK[j], wKp[j]);
  }
  pack_fuse_k<<<(65 * 288 + 255) / 256, 256, 0, stream>>>((const float*)d_in[37], ws + oFUSE);

  // ---- build fbt ----
  {
    int n = 8 * 65 * (int)HW1;
    build_fbt_k<<<(n + 255) / 256, 256, 0, stream>>>(features, tptr, ws + oFBT);
  }

  // 3x3 conv, LDS-staged
  auto conv3 = [&](size_t actOff, size_t wOff, int bi, size_t outOff,
                   int Nimg, int Cin, int Hin, int Win, int Cout, int Hout, int Wout,
                   int Kp, int Ktot, int lrelu) {
    unsigned grid = (unsigned)(Nimg * Hout * (Wout >> 7));
    size_t smem = (size_t)Cin * 396 * sizeof(_Float16);
    conv_lds_k<0><<<grid, 256, smem, stream>>>(ws + actOff, ws + wOff,
        (const float*)d_in[bi], ws + outOff,
        Nimg, Cin, Hin, Win, Cout, Hout, Wout, Kp, Ktot, lrelu);
  };
  // convT, direct
  auto convt = [&](size_t actOff, size_t wOff, int bi, size_t outOff,
                   int Nimg, int Cin, int Hin, int Win, int Cout, int Hout, int Wout,
                   int Kp, int Ktot) {
    long long waves = (long long)((Cout + 15) / 16) * (long long)Nimg *
                      ((long long)Hout * Wout / 16);
    unsigned grid = (unsigned)((waves * 32 + 255) / 256);
    convt_wmma_k<<<grid, 256, 0, stream>>>(ws + actOff, ws + wOff,
        (const float*)d_in[bi], ws + outOff,
        Nimg, Cin, Hin, Win, Cout, Hout, Wout, Kp, Ktot, 0);
  };

  // ---- fuse 1x1 over [2][260][128][128] view of fbt -> occ_feat (async LDS) ----
  {
    unsigned grid = 2 * 128 * 1;
    size_t smem = 260ull * 128 * sizeof(_Float16);
    conv_lds_k<1><<<grid, 256, smem, stream>>>(ws + oFBT, ws + oFUSE,
        (const float*)d_in[38], ws + oOCF,
        2, 260, 128, 128, 65, 128, 128, 288, 260, 1);
  }

  // ---- kernel-prediction subnets ----
  for (int p = 0; p < 3; ++p) {
    int sw = p * 4;
    conv3(oFBT, woff[sw + 0], bidx[sw + 0], oT1,     8, 65, 128, 128, 64, 128, 128, 608, 585, 1);
    conv3(oT1,  woff[sw + 1], bidx[sw + 1], oT2,     8, 64, 128, 128, 25, 128, 128, 576, 576, 1);
    convt(oT2,  woff[sw + 2], bidx[sw + 2], oT3,     8, 25, 128, 128, 25, 256, 256, 256, 225);
    conv3(oT3,  woff[sw + 3], bidx[sw + 3], oBIG[p], 8, 25, 256, 256, 25, 256, 256, 256, 225, 0);
  }
  softmax_ch_k<25><<<(8 * (int)HW2 + 255) / 256, 256, 0, stream>>>(ws + oBIG[0], 8, (int)HW2);

  // ---- occlusion subnet ----
  conv3(oOCF, woff[12], bidx[12], oT1,  2, 65, 128, 128, 64, 128, 128, 608, 585, 1);
  conv3(oT1,  woff[13], bidx[13], oT2,  2, 64, 128, 128, 64, 128, 128, 576, 576, 1);
  convt(oT2,  woff[14], bidx[14], oT3,  2, 64, 128, 128, 64, 256, 256, 576, 576);
  conv3(oT3,  woff[15], bidx[15], oOCL, 2, 64, 256, 256, 4,  256, 256, 576, 576, 0);
  softmax_ch_k<4><<<(2 * (int)HW2 + 255) / 256, 256, 0, stream>>>(ws + oOCL, 2, (int)HW2);

  // ---- synthesis ----
  synth_k<<<(2 * (int)HW2 + 255) / 256, 256, 0, stream>>>(
      ws + oBIG[0], ws + oBIG[1], ws + oBIG[2], ws + oOCL, frames, (float*)d_out);
}
